// NliClassifier_48284022341796
// MI455X (gfx1250) — compile-verified, matrix operand order checked by reading, not run on hardware
//
#include <hip/hip_runtime.h>
#include <hip/hip_bf16.h>
#include <math.h>
#include <stdint.h>

// ---------------------------------------------------------------------------
// NLI siamese-LSTM classifier for gfx1250 (MI455X).
//
//  * LSTM recurrence: per-step [32 x 384] x [384 x 1024] GEMM with
//    v_wmma_f32_16x16x32_f16, K = concat(x_t (128) | h_{t-1} (256)).
//  * One workgroup (16 waves / 512 threads) owns 32 batch rows for all 256
//    timesteps of one sequence -> grid (16, 2).
//  * Wave w owns hidden columns [16w,16w+16): i/f/g/o tiles for both M-tiles;
//    LSTM cell state c lives in 16 VGPRs/lane.
//  * Weights pre-swizzled to the CDNA5 16-bit B-matrix (32x16) layout and
//    re-streamed from L2 each step through an address_space(1) laundered
//    pointer -> global_load_b128 (LOADcnt only), no LICM scratch spills.
//  * Embedding table pre-converted to f16; per-step x rows staged into LDS
//    with global_load_async_to_lds_b128 (ASYNCcnt), double-buffered so the
//    t+1 gather overlaps the step-t GEMM. Fallback: synchronous copy.
//  * LDS row layout [x0(128) | h(256) | x1(128) | pad], pitch 528 halves
//    (264 dwords = 8 mod 64 banks) -> A-fragment ds_load_b128 conflict-free.
// ---------------------------------------------------------------------------

typedef _Float16 v16h __attribute__((ext_vector_type(16)));
typedef _Float16 v8h  __attribute__((ext_vector_type(8)));
typedef _Float16 v4h  __attribute__((ext_vector_type(4)));
typedef float    v8f  __attribute__((ext_vector_type(8)));
typedef int      v4i  __attribute__((vector_size(4 * sizeof(int))));

#define GLOBAL_AS __attribute__((address_space(1)))
#define LDS_AS    __attribute__((address_space(3)))

#define VOCAB 50000
#define BATCH 512
#define SEQ   256
#define EMB   128
#define HID   256
#define GDIM  1024          // 4*HID
#define KTOT  384           // EMB + HID
#define KCH   12            // KTOT / 32
#define BM    32            // batch rows per workgroup (2 M-tiles)
#define MT    2             // M-tiles per workgroup
#define NTILES 64           // GDIM / 16
#define XPITCH 528          // LDS row pitch in halves: [x0|h|x1|pad], 8 mod 64 banks
#define HOFF   128          // h region column offset (== K offset of h)
#define X1OFF  384          // second x buffer column offset

// workspace layout (bytes)
#define WS_WF    0                         // f16 swizzled weights: 786432
#define WS_BSUM  786432                    // f32 fused bias: 4096
#define WS_HBUF  790528                    // f32 hT: 2*512*256*4 = 1048576
#define WS_EMBF  1839104                   // f16 embedding: 50000*128*2 = 12800000

#if defined(__has_builtin)
#if __has_builtin(__builtin_amdgcn_global_load_async_to_lds_b128) && \
    __has_builtin(__builtin_amdgcn_s_wait_asynccnt)
#define USE_ASYNC_LDS 1
#endif
#endif

// --------------------------- prep kernels ----------------------------------
// Wf[((nt*KCH + kc)*32 + lane)*16 + i] = f16( W[k][n] )
//   n = nt*16 + (lane&15);  k = kc*32 + (i/8)*16 + (lane>>4)*8 + (i&7)
__global__ __launch_bounds__(256) void prep_wf_kernel(const float* __restrict__ w_ih,
                                                      const float* __restrict__ w_hh,
                                                      _Float16* __restrict__ Wf) {
    int idx = blockIdx.x * 256 + threadIdx.x;
    if (idx >= NTILES * KCH * 32 * 16) return;
    int i    = idx & 15;
    int lane = (idx >> 4) & 31;
    int kc   = (idx >> 9) % KCH;
    int nt   = idx / (KCH * 512);
    int n = nt * 16 + (lane & 15);
    int k = kc * 32 + ((i >> 3) << 4) + ((lane >> 4) << 3) + (i & 7);
    float v = (k < EMB) ? w_ih[n * EMB + k] : w_hh[n * HID + (k - EMB)];
    Wf[idx] = (_Float16)v;
}

__global__ __launch_bounds__(256) void prep_bias_kernel(const float* __restrict__ b_ih,
                                                        const float* __restrict__ b_hh,
                                                        float* __restrict__ bsum) {
    int n = blockIdx.x * 256 + threadIdx.x;
    if (n < GDIM) bsum[n] = b_ih[n] + b_hh[n];
}

__global__ __launch_bounds__(256) void prep_emb_kernel(const float* __restrict__ emb,
                                                       _Float16* __restrict__ embf) {
    int idx = blockIdx.x * 256 + threadIdx.x;          // one per 4 elements
    if (idx >= VOCAB * EMB / 4) return;
    const float4 v = ((const float4*)emb)[idx];
    v4h p;
    p.x = (_Float16)v.x; p.y = (_Float16)v.y;
    p.z = (_Float16)v.z; p.w = (_Float16)v.w;
    ((v4h*)embf)[idx] = p;
}

// ------------------------------- LSTM core ---------------------------------
__device__ __forceinline__ float sigf(float x) { return 1.0f / (1.0f + __expf(-x)); }

__global__ __launch_bounds__(512)
void lstm_kernel(const int* __restrict__ s1, const int* __restrict__ s2,
                 const _Float16* __restrict__ embf,
                 const _Float16* __restrict__ Wf,
                 const float* __restrict__ bsum,
                 float* __restrict__ hbuf) {
    __shared__ __align__(16) _Float16 xh[BM * XPITCH];   // 33 KB

    const int mtile0 = blockIdx.x;           // 0..15
    const int seq    = blockIdx.y;           // 0..1
    const int* __restrict__ s = seq ? s2 : s1;
    const int bm0  = mtile0 * BM;
    const int tid  = threadIdx.x;
    const int wave = tid >> 5;               // 0..15
    const int lane = tid & 31;
    const int nl   = lane & 15;
    const int hi   = lane >> 4;

    // zero the h region of xh (h_0 = 0)
    for (int idx = tid; idx < BM * HID; idx += 512) {
        xh[(idx >> 8) * XPITCH + HOFF + (idx & 255)] = (_Float16)0.0f;
    }

    const int ncol = wave * 16 + nl;         // hidden column in [0,256)
    const float bi = bsum[0 * HID + ncol];
    const float bf = bsum[1 * HID + ncol];
    const float bg = bsum[2 * HID + ncol];
    const float bo = bsum[3 * HID + ncol];

    const v8f vzero = {};
    v8f c[MT];     c[0] = vzero; c[1] = vzero;
    v8f hlast[MT]; hlast[0] = vzero; hlast[1] = vzero;

    // x-stage mapping: thread -> (row, 16-byte segment)
    const int xrow = tid >> 4;               // 0..31
    const int xcg  = (tid & 15) << 3;        // half offset 0,8,...,120

    uintptr_t wfbase = (uintptr_t)Wf;

    // prologue: stage x_0 into buffer 0
    {
        int tok = s[(bm0 + xrow) * SEQ + 0];
        const _Float16* src = embf + (size_t)tok * EMB + xcg;
        _Float16* dst = &xh[xrow * XPITCH + xcg];
#if defined(USE_ASYNC_LDS)
        __builtin_amdgcn_global_load_async_to_lds_b128(
            (GLOBAL_AS v4i*)src, (LDS_AS v4i*)dst, 0, 0);
#else
        *(v8h*)dst = *(const v8h*)src;
#endif
    }

    for (int t = 0; t < SEQ; ++t) {
        const int xsel = t & 1;
        // launder weight base each step: keeps B loads as in-loop global
        // loads from L2 instead of LICM-hoisted scratch spills.
        asm volatile("" : "+s"(wfbase));
        const GLOBAL_AS _Float16* wf_t = (const GLOBAL_AS _Float16*)wfbase;

#if defined(USE_ASYNC_LDS)
        __builtin_amdgcn_s_wait_asynccnt(0);   // x_t landed in LDS
#endif
        __syncthreads();   // x_t + h_{t-1} visible to all waves

        // stage x_{t+1} into the other buffer; overlaps this step's GEMM
        if (t + 1 < SEQ) {
            int tok = s[(bm0 + xrow) * SEQ + t + 1];
            const _Float16* src = embf + (size_t)tok * EMB + xcg;
            _Float16* dst = &xh[xrow * XPITCH + (xsel ? 0 : X1OFF) + xcg];
#if defined(USE_ASYNC_LDS)
            __builtin_amdgcn_global_load_async_to_lds_b128(
                (GLOBAL_AS v4i*)src, (LDS_AS v4i*)dst, 0, 0);
#else
            *(v8h*)dst = *(const v8h*)src;
#endif
        }

        // ---- gates = [x|h] @ W^T : 2 M-tiles x 4 gate tiles, K = 384 ----
        v8f acc[MT][4];
#pragma unroll
        for (int mt = 0; mt < MT; ++mt)
#pragma unroll
            for (int g = 0; g < 4; ++g) acc[mt][g] = vzero;

        const int xoff = xsel ? X1OFF : 0;   // column base of current x buffer
#pragma unroll
        for (int kc = 0; kc < KCH; ++kc) {
            // kc 0..3 -> x region (K 0..127), kc 4..11 -> h region (K 128..383)
            const int colb = (kc < 4) ? (kc * 32 + xoff) : (kc * 32 + (HOFF - EMB));
            const int kbase = colb + (hi << 3);
            v16h a[MT];
#pragma unroll
            for (int mt = 0; mt < MT; ++mt) {
                const int arow = mt * 16 + nl;
                v8h lo = *(const v8h*)(&xh[arow * XPITCH + kbase]);
                v8h ph = *(const v8h*)(&xh[arow * XPITCH + kbase + 16]);
                a[mt] = __builtin_shufflevector(lo, ph, 0, 1, 2, 3, 4, 5, 6, 7,
                                                        8, 9, 10, 11, 12, 13, 14, 15);
            }
            const GLOBAL_AS _Float16* wb =
                wf_t + ((size_t)(wave * KCH + kc) * 32 + lane) * 16;
            // gate g tile index nt = g*16 + wave -> stride 16*KCH*512 halves
            v16h b0 = *(const GLOBAL_AS v16h*)(wb + (size_t)0 * 16 * KCH * 512);
            v16h b1 = *(const GLOBAL_AS v16h*)(wb + (size_t)1 * 16 * KCH * 512);
            v16h b2 = *(const GLOBAL_AS v16h*)(wb + (size_t)2 * 16 * KCH * 512);
            v16h b3 = *(const GLOBAL_AS v16h*)(wb + (size_t)3 * 16 * KCH * 512);
#pragma unroll
            for (int mt = 0; mt < MT; ++mt) {
                acc[mt][0] = __builtin_amdgcn_wmma_f32_16x16x32_f16(
                    false, a[mt], false, b0, (short)0, acc[mt][0], false, false);
                acc[mt][1] = __builtin_amdgcn_wmma_f32_16x16x32_f16(
                    false, a[mt], false, b1, (short)0, acc[mt][1], false, false);
                acc[mt][2] = __builtin_amdgcn_wmma_f32_16x16x32_f16(
                    false, a[mt], false, b2, (short)0, acc[mt][2], false, false);
                acc[mt][3] = __builtin_amdgcn_wmma_f32_16x16x32_f16(
                    false, a[mt], false, b3, (short)0, acc[mt][3], false, false);
            }
        }
        __syncthreads();   // all WMMA LDS reads done before h overwrite

        // ---- LSTM cell update; C/D layout: VGPR r -> row r+8*hi, col ncol ----
#pragma unroll
        for (int mt = 0; mt < MT; ++mt) {
#pragma unroll
            for (int r = 0; r < 8; ++r) {
                float gi = sigf(acc[mt][0][r] + bi);
                float gf = sigf(acc[mt][1][r] + bf);
                float gg = tanhf(acc[mt][2][r] + bg);
                float go = sigf(acc[mt][3][r] + bo);
                float cn = gf * c[mt][r] + gi * gg;
                c[mt][r] = cn;
                float hn = go * tanhf(cn);
                hlast[mt][r] = hn;
                xh[(mt * 16 + r + (hi << 3)) * XPITCH + HOFF + ncol] = (_Float16)hn;
            }
        }
        // next iteration's first barrier orders these h stores vs. WMMA reads
    }

    // ---- write hT (fp32) ----
#pragma unroll
    for (int mt = 0; mt < MT; ++mt)
#pragma unroll
        for (int r = 0; r < 8; ++r) {
            int row = mt * 16 + r + (hi << 3);
            hbuf[(size_t)(seq * BATCH + bm0 + row) * HID + ncol] = hlast[mt][r];
        }
}

// ------------------------------- MLP head ----------------------------------
__global__ __launch_bounds__(256)
void head_kernel(const float* __restrict__ hbuf,
                 const float* __restrict__ w_hid, const float* __restrict__ b_hid,
                 const float* __restrict__ w_out, const float* __restrict__ b_out,
                 float* __restrict__ out) {
    __shared__ float cat[2 * HID];
    __shared__ float hid[HID];
    __shared__ float red[HID];
    const int b = blockIdx.x;
    const int n = threadIdx.x;

    cat[n]       = hbuf[(size_t)b * HID + n];                 // h1
    cat[HID + n] = hbuf[(size_t)(BATCH + b) * HID + n];       // h2
    __syncthreads();

    float acc = b_hid[n];
    const float* wrow = w_hid + (size_t)n * 2 * HID;
#pragma unroll 8
    for (int k = 0; k < 2 * HID; ++k) acc = fmaf(cat[k], wrow[k], acc);
    hid[n] = fmaxf(acc, 0.0f);
    __syncthreads();

    for (int o = 0; o < 3; ++o) {
        red[n] = hid[n] * w_out[o * HID + n];
        __syncthreads();
        for (int s2 = HID / 2; s2 > 0; s2 >>= 1) {
            if (n < s2) red[n] += red[n + s2];
            __syncthreads();
        }
        if (n == 0) out[b * 3 + o] = red[0] + b_out[o];
        __syncthreads();
    }
}

// ------------------------------- launcher ----------------------------------
extern "C" void kernel_launch(void* const* d_in, const int* in_sizes, int n_in,
                              void* d_out, int out_size, void* d_ws, size_t ws_size,
                              hipStream_t stream) {
    const int*   s1    = (const int*)d_in[0];
    const int*   s2    = (const int*)d_in[1];
    const float* emb   = (const float*)d_in[2];
    const float* w_ih  = (const float*)d_in[3];
    const float* w_hh  = (const float*)d_in[4];
    const float* b_ih  = (const float*)d_in[5];
    const float* b_hh  = (const float*)d_in[6];
    const float* w_hid = (const float*)d_in[7];
    const float* b_hid = (const float*)d_in[8];
    const float* w_out = (const float*)d_in[9];
    const float* b_out = (const float*)d_in[10];

    char* ws = (char*)d_ws;
    _Float16* Wf    = (_Float16*)(ws + WS_WF);
    float*    bsum  = (float*)(ws + WS_BSUM);
    float*    hbuf  = (float*)(ws + WS_HBUF);
    _Float16* embf  = (_Float16*)(ws + WS_EMBF);

    prep_wf_kernel<<<393216 / 256, 256, 0, stream>>>(w_ih, w_hh, Wf);
    prep_bias_kernel<<<GDIM / 256, 256, 0, stream>>>(b_ih, b_hh, bsum);
    prep_emb_kernel<<<(VOCAB * EMB / 4 + 255) / 256, 256, 0, stream>>>(emb, embf);

    dim3 grid(BATCH / BM, 2);
    lstm_kernel<<<grid, 512, 0, stream>>>(s1, s2, embf, Wf, bsum, hbuf);

    head_kernel<<<BATCH, 256, 0, stream>>>(hbuf, w_hid, b_hid, w_out, b_out,
                                           (float*)d_out);
}
